// Model_70042326663581
// MI455X (gfx1250) — matrix operand order repaired and checked by reading, not compiled, not run on hardware
//
#include <hip/hip_runtime.h>
#include <hip/hip_bf16.h>
#include <math.h>

typedef __bf16 bf16_t;
typedef bf16_t v2bf  __attribute__((ext_vector_type(2)));
typedef bf16_t v4bf  __attribute__((ext_vector_type(4)));
typedef bf16_t v8bf  __attribute__((ext_vector_type(8)));
typedef bf16_t v16bf __attribute__((ext_vector_type(16)));
typedef float  v8f   __attribute__((ext_vector_type(8)));
typedef unsigned int v4u __attribute__((ext_vector_type(4)));
typedef unsigned int v8u __attribute__((ext_vector_type(8)));

#define BTOK 2048
#define DDIM 1024
#define FDIM 4096
#define NEXP 16
#define NGEN 2
#define NALL 18

#define BM 128
#define BN 128
#define BK 32
#define BKP 40   // padded k-stride for transposed B tile

// shared memory: two A bufs (BM*BK bf16) + two B bufs (BN*BKP bf16) = 36864 B;
// FC1 epilogue reuses the same block as a BM*BN bf16 staging tile (32768 B).
#define SMEM_BYTES (2 * BM * BK * 2 + 2 * BN * BKP * 2)

__device__ __forceinline__ float gelu_exact(float v) {
    return 0.5f * v * (1.0f + erff(v * 0.70710678118654752f));
}

__device__ __forceinline__ v16bf cat8(v8bf lo, v8bf hi) {
    return __builtin_shufflevector(lo, hi, 0,1,2,3,4,5,6,7,8,9,10,11,12,13,14,15);
}

__device__ __forceinline__ float f4c(const float4& v, int j) {
    switch (j) { case 0: return v.x; case 1: return v.y; case 2: return v.z; default: return v.w; }
}

// ---------------------------------------------------------------------------
// TDM: 2-D tile load Global->LDS.  tile = 128 rows x 32 bf16 (64 B/row),
// tensor row stride = FDIM elements.  Issued once per wave; EXEC ignored.
// D# bitfields per CDNA5 ISA 8.3/8.4.
// ---------------------------------------------------------------------------
__device__ __forceinline__ void tdm_load_tile_bf16(unsigned lds_off, const bf16_t* gptr)
{
    const unsigned long long ga = (unsigned long long)(size_t)(const void*)gptr;
    v4u g0;
    g0.x = 1u;                                                  // count=1 (valid), user mode
    g0.y = lds_off;                                             // lds_addr (bytes)
    g0.z = (unsigned)ga;                                        // global_addr[31:0]
    g0.w = (unsigned)((ga >> 32) & 0x01FFFFFFu) | (2u << 30);   // addr[56:32] | type=2
    v8u g1;
    g1.s0 = (1u << 16);                          // workgroup_mask=0, data_size=1 (2 B)
    g1.s1 = ((unsigned)FDIM & 0xFFFFu) << 16;    // tensor_dim0 lo16 (4096)
    g1.s2 = ((unsigned)FDIM >> 16) | (((unsigned)BTOK & 0xFFFFu) << 16); // td0 hi | td1 lo
    g1.s3 = ((unsigned)BTOK >> 16) | (32u << 16);               // td1 hi | tile_dim0=32
    g1.s4 = 128u;                                // tile_dim1=128, tile_dim2=0
    g1.s5 = (unsigned)FDIM;                      // tensor_dim0_stride lo32
    g1.s6 = 0u;                                  // stride0 hi16 | stride1 lo16
    g1.s7 = 0u;
    asm volatile("tensor_load_to_lds %0, %1" : : "s"(g0), "s"(g1) : "memory");
}

// ---------------------------------------------------------------------------
// Gating + loss partials
// ---------------------------------------------------------------------------
__global__ void gate_kernel(const float* __restrict__ logits,
                            const int* __restrict__ masks,
                            float* __restrict__ gates,
                            float* __restrict__ lpart)
{
    const int b = blockIdx.x * blockDim.x + threadIdx.x;
    if (b >= BTOK) return;
    float raw[NEXP];
    float mx = -1e30f;
    #pragma unroll
    for (int e = 0; e < NEXP; ++e) { raw[e] = logits[b * NEXP + e]; mx = fmaxf(mx, raw[e]); }
    float s = 0.f;
    #pragma unroll
    for (int e = 0; e < NEXP; ++e) { raw[e] = __expf(raw[e] - mx); s += raw[e]; }
    const float inv = 1.f / s;
    float g[NEXP];
    float gsum = 0.f;
    #pragma unroll
    for (int e = 0; e < NEXP; ++e) {
        raw[e] *= inv;
        g[e] = (masks[b * NEXP + e] == 1) ? raw[e] : 0.f;
        gsum += g[e];
    }
    const float inv2 = 1.f / (gsum + 1e-9f);
    #pragma unroll
    for (int e = 0; e < NEXP; ++e) {
        const float ge = g[e] * inv2;
        gates[b * NEXP + e] = ge;
        atomicAdd(&lpart[e], raw[e]);
        if (ge != 0.f) atomicAdd(&lpart[NEXP + e], 1.0f);
    }
    atomicAdd(&lpart[2 * NEXP], gsum);
}

// ---------------------------------------------------------------------------
// FC1: H[e] = gelu(x @ Win[e]).  f32->bf16 in regs, double-buffered LDS,
// WMMA bf16, LDS-staged coalesced epilogue.
// grid = (F/BN, B/BM, 18), block = 256 (8 waves, 2x4 over 128x128 tile)
// ---------------------------------------------------------------------------
__global__ __launch_bounds__(256)
void fc1_gelu_kernel(const float* __restrict__ x,
                     const float* __restrict__ ew_in,
                     const float* __restrict__ gw_in,
                     bf16_t* __restrict__ H)
{
    __shared__ __align__(16) unsigned char smem[SMEM_BYTES];
    bf16_t* AsBase = (bf16_t*)smem;                         // 2 x BM*BK
    bf16_t* BsBase = (bf16_t*)(smem + 2 * BM * BK * 2);     // 2 x BN*BKP

    const int e = blockIdx.z;
    const float* __restrict__ Win = (e < NEXP)
        ? (ew_in + (size_t)e * DDIM * FDIM)
        : (gw_in + (size_t)(e - NEXP) * DDIM * FDIM);

    const int tile_n = blockIdx.x * BN;
    const int tile_m = blockIdx.y * BM;
    const int tid  = threadIdx.x;
    const int lane = tid & 31;
    const int wave = tid >> 5;
    const int wm = wave & 1;
    const int wn = wave >> 1;
    const int lrow = lane & 15;
    const int ksel = lane >> 4;

    // fill-thread mapping
    const int a_kc = (tid & 7) * 4;           // A: 4 consecutive k
    const int a_r0 = tid >> 3;                // A: row 0..31 (x4 groups of 32)
    const int b_nc = (tid & 31) * 4;          // B: 4 consecutive n
    const int b_kr = (tid >> 5) * 2;          // B: k-pair base (0,2,..,14; +16 for 2nd half)

    const v8f zero8 = {0.f,0.f,0.f,0.f,0.f,0.f,0.f,0.f};
    v8f acc[4][2];
    #pragma unroll
    for (int mt = 0; mt < 4; ++mt)
        #pragma unroll
        for (int nt = 0; nt < 2; ++nt) acc[mt][nt] = zero8;

    float4 a_reg[4], b_reg[4];

    // ---- stage tile k0 into registers (all global loads issued together) ----
    auto load_regs = [&](int k0) {
        #pragma unroll
        for (int rr = 0; rr < 4; ++rr)
            a_reg[rr] = *(const float4*)(x + (size_t)(tile_m + a_r0 + rr * 32) * DDIM + (k0 + a_kc));
        #pragma unroll
        for (int kk = 0; kk < 2; ++kk) {
            b_reg[2*kk+0] = *(const float4*)(Win + (size_t)(k0 + b_kr + 16*kk)     * FDIM + (tile_n + b_nc));
            b_reg[2*kk+1] = *(const float4*)(Win + (size_t)(k0 + b_kr + 16*kk + 1) * FDIM + (tile_n + b_nc));
        }
    };
    // ---- convert + store registers into LDS buffer `buf` ----
    auto store_lds = [&](int buf) {
        bf16_t* Asb = AsBase + buf * BM * BK;
        bf16_t* Bsb = BsBase + buf * BN * BKP;
        #pragma unroll
        for (int rr = 0; rr < 4; ++rr) {
            v4bf p;
            p.x = (bf16_t)a_reg[rr].x; p.y = (bf16_t)a_reg[rr].y;
            p.z = (bf16_t)a_reg[rr].z; p.w = (bf16_t)a_reg[rr].w;
            *(v4bf*)(&Asb[(a_r0 + rr * 32) * BK + a_kc]) = p;
        }
        #pragma unroll
        for (int kk = 0; kk < 2; ++kk)
            #pragma unroll
            for (int j = 0; j < 4; ++j) {
                v2bf p;
                p.x = (bf16_t)f4c(b_reg[2*kk+0], j);
                p.y = (bf16_t)f4c(b_reg[2*kk+1], j);
                *(v2bf*)(&Bsb[(b_nc + j) * BKP + (b_kr + 16 * kk)]) = p;
            }
    };
    // ---- 8 WMMAs from LDS buffer `buf` ----
    auto compute = [&](int buf) {
        const bf16_t* Asb = AsBase + buf * BM * BK;
        const bf16_t* Bsb = BsBase + buf * BN * BKP;
        v16bf afrag[4];
        #pragma unroll
        for (int mt = 0; mt < 4; ++mt) {
            const bf16_t* ap = &Asb[(wm * 64 + mt * 16 + lrow) * BK + ksel * 8];
            afrag[mt] = cat8(*(const v8bf*)ap, *(const v8bf*)(ap + 16));
        }
        v16bf bfrag[2];
        #pragma unroll
        for (int nt = 0; nt < 2; ++nt) {
            const bf16_t* bp = &Bsb[(wn * 32 + nt * 16 + lrow) * BKP + ksel * 8];
            bfrag[nt] = cat8(*(const v8bf*)bp, *(const v8bf*)(bp + 16));
        }
        #pragma unroll
        for (int mt = 0; mt < 4; ++mt)
            #pragma unroll
            for (int nt = 0; nt < 2; ++nt)
                acc[mt][nt] = __builtin_amdgcn_wmma_f32_16x16x32_bf16(
                    false, afrag[mt], false, bfrag[nt], (short)0, acc[mt][nt], false, false);
    };

    // ---- software pipeline: 1 barrier per k-step ----
    load_regs(0);
    store_lds(0);
    __syncthreads();
    int cur = 0;
    for (int k0 = BK; k0 <= DDIM; k0 += BK) {
        const bool last = (k0 == DDIM);
        if (!last) load_regs(k0);
        compute(cur);
        if (!last) store_lds(cur ^ 1);
        __syncthreads();
        cur ^= 1;
    }

    // ---- epilogue: gelu -> LDS C-tile -> coalesced b128 stores ----
    bf16_t* Ct = (bf16_t*)smem;   // BM x BN bf16 (buffers are dead now)
    #pragma unroll
    for (int mt = 0; mt < 4; ++mt) {
        const int rl = wm * 64 + mt * 16 + ksel * 8;
        #pragma unroll
        for (int nt = 0; nt < 2; ++nt) {
            const int cl = wn * 32 + nt * 16 + lrow;
            #pragma unroll
            for (int i = 0; i < 8; ++i)
                Ct[(rl + i) * BN + cl] = (bf16_t)gelu_exact(acc[mt][nt][i]);
        }
    }
    __syncthreads();
    bf16_t* __restrict__ Hout = H + (size_t)e * BTOK * FDIM;
    const int crow = tid >> 1;
    const int ccol = (tid & 1) * 64;
    bf16_t* dst = Hout + (size_t)(tile_m + crow) * FDIM + (tile_n + ccol);
    const bf16_t* src = &Ct[crow * BN + ccol];
    #pragma unroll
    for (int j = 0; j < 8; ++j)
        *(v8bf*)(dst + j * 8) = *(const v8bf*)(src + j * 8);
}

// ---------------------------------------------------------------------------
// FC2: acc += scale(e,b) * (H[e] @ Wout[e] + bias[e]).
// A-tile (bf16 copy of H) moved by the Tensor Data Mover; B-tile converted
// f32->bf16 in regs.  Double-buffered, 1 barrier per k-step.
// grid = (D/BN, B/BM, 18)
// ---------------------------------------------------------------------------
__global__ __launch_bounds__(256)
void fc2_combine_kernel(const bf16_t* __restrict__ H,
                        const float* __restrict__ ew_out,
                        const float* __restrict__ eb_out,
                        const float* __restrict__ gw_out,
                        const float* __restrict__ gb_out,
                        const float* __restrict__ gates,
                        float* __restrict__ accbuf)
{
    __shared__ __align__(16) unsigned char smem[SMEM_BYTES];
    bf16_t* AsBase = (bf16_t*)smem;
    bf16_t* BsBase = (bf16_t*)(smem + 2 * BM * BK * 2);
    const unsigned ldsA0 = (unsigned)(size_t)(void*)AsBase;

    const int e = blockIdx.z;
    const float* __restrict__ Wout = (e < NEXP)
        ? (ew_out + (size_t)e * FDIM * DDIM)
        : (gw_out + (size_t)(e - NEXP) * FDIM * DDIM);
    const bf16_t* __restrict__ Hin = H + (size_t)e * BTOK * FDIM;

    const int tile_n = blockIdx.x * BN;   // over D
    const int tile_m = blockIdx.y * BM;   // over B
    const int tid  = threadIdx.x;
    const int lane = tid & 31;
    const int wave = tid >> 5;
    const int wm = wave & 1;
    const int wn = wave >> 1;
    const int lrow = lane & 15;
    const int ksel = lane >> 4;

    const int b_nc = (tid & 31) * 4;
    const int b_kr = (tid >> 5) * 2;

    const v8f zero8 = {0.f,0.f,0.f,0.f,0.f,0.f,0.f,0.f};
    v8f acc[4][2];
    #pragma unroll
    for (int mt = 0; mt < 4; ++mt)
        #pragma unroll
        for (int nt = 0; nt < 2; ++nt) acc[mt][nt] = zero8;

    float4 b_reg[4];
    auto load_b_regs = [&](int k0) {
        #pragma unroll
        for (int kk = 0; kk < 2; ++kk) {
            b_reg[2*kk+0] = *(const float4*)(Wout + (size_t)(k0 + b_kr + 16*kk)     * DDIM + (tile_n + b_nc));
            b_reg[2*kk+1] = *(const float4*)(Wout + (size_t)(k0 + b_kr + 16*kk + 1) * DDIM + (tile_n + b_nc));
        }
    };
    auto store_b_lds = [&](int buf) {
        bf16_t* Bsb = BsBase + buf * BN * BKP;
        #pragma unroll
        for (int kk = 0; kk < 2; ++kk)
            #pragma unroll
            for (int j = 0; j < 4; ++j) {
                v2bf p;
                p.x = (bf16_t)f4c(b_reg[2*kk+0], j);
                p.y = (bf16_t)f4c(b_reg[2*kk+1], j);
                *(v2bf*)(&Bsb[(b_nc + j) * BKP + (b_kr + 16 * kk)]) = p;
            }
    };
    auto compute = [&](int buf) {
        const bf16_t* Asb = AsBase + buf * BM * BK;
        const bf16_t* Bsb = BsBase + buf * BN * BKP;
        v16bf afrag[4];
        #pragma unroll
        for (int mt = 0; mt < 4; ++mt) {
            const bf16_t* ap = &Asb[(wm * 64 + mt * 16 + lrow) * BK + ksel * 8];
            afrag[mt] = cat8(*(const v8bf*)ap, *(const v8bf*)(ap + 16));
        }
        v16bf bfrag[2];
        #pragma unroll
        for (int nt = 0; nt < 2; ++nt) {
            const bf16_t* bp = &Bsb[(wn * 32 + nt * 16 + lrow) * BKP + ksel * 8];
            bfrag[nt] = cat8(*(const v8bf*)bp, *(const v8bf*)(bp + 16));
        }
        #pragma unroll
        for (int mt = 0; mt < 4; ++mt)
            #pragma unroll
            for (int nt = 0; nt < 2; ++nt)
                acc[mt][nt] = __builtin_amdgcn_wmma_f32_16x16x32_bf16(
                    false, afrag[mt], false, bfrag[nt], (short)0, acc[mt][nt], false, false);
    };

    // ---- pipeline with TDM on the A path ----
    if (wave == 0)
        tdm_load_tile_bf16(ldsA0, Hin + (size_t)tile_m * FDIM);
    load_b_regs(0);
    store_b_lds(0);
    if (wave == 0) __builtin_amdgcn_s_wait_tensorcnt(0);
    __syncthreads();
    int cur = 0;
    for (int k0 = BK; k0 <= FDIM; k0 += BK) {
        const bool last = (k0 == FDIM);
        if (!last) {
            if (wave == 0)
                tdm_load_tile_bf16(ldsA0 + (unsigned)((cur ^ 1) * BM * BK * 2),
                                   Hin + (size_t)tile_m * FDIM + k0);
            load_b_regs(k0);
        }
        compute(cur);
        if (!last) {
            store_b_lds(cur ^ 1);
            if (wave == 0) __builtin_amdgcn_s_wait_tensorcnt(0);
        }
        __syncthreads();
        cur ^= 1;
    }

    // ---- epilogue: bias, gate, cross-expert accumulate ----
    #pragma unroll
    for (int mt = 0; mt < 4; ++mt) {
        const int rbase = tile_m + wm * 64 + mt * 16 + ksel * 8;
        #pragma unroll
        for (int nt = 0; nt < 2; ++nt) {
            const int col = tile_n + wn * 32 + nt * 16 + lrow;
            const float bias = (e < NEXP) ? eb_out[e * DDIM + col]
                                          : gb_out[(e - NEXP) * DDIM + col];
            #pragma unroll
            for (int i = 0; i < 8; ++i) {
                const int row = rbase + i;
                float v = acc[mt][nt][i] + bias;
                if (e < NEXP) v *= gates[row * NEXP + e];
                atomicAdd(&accbuf[(size_t)row * DDIM + col], v);
            }
        }
    }
}

// ---------------------------------------------------------------------------
// Residual + LayerNorm
// ---------------------------------------------------------------------------
__global__ __launch_bounds__(256)
void ln_kernel(const float* __restrict__ accbuf, const float* __restrict__ x,
               const float* __restrict__ gamma, const float* __restrict__ beta,
               float* __restrict__ out)
{
    const int b = blockIdx.x;
    const int tid = threadIdx.x;
    __shared__ float red[16];
    float y[4];
    float s = 0.f, s2 = 0.f;
    #pragma unroll
    for (int j = 0; j < 4; ++j) {
        const int d = tid + j * 256;
        y[j] = accbuf[(size_t)b * DDIM + d] + x[(size_t)b * DDIM + d];
        s += y[j]; s2 += y[j] * y[j];
    }
    #pragma unroll
    for (int o = 16; o > 0; o >>= 1) {
        s  += __shfl_xor(s,  o, 32);
        s2 += __shfl_xor(s2, o, 32);
    }
    const int w = tid >> 5;
    if ((tid & 31) == 0) { red[w] = s; red[8 + w] = s2; }
    __syncthreads();
    float ts = 0.f, ts2 = 0.f;
    #pragma unroll
    for (int i = 0; i < 8; ++i) { ts += red[i]; ts2 += red[8 + i]; }
    const float mu   = ts / DDIM;
    const float var  = ts2 / DDIM - mu * mu;
    const float rstd = rsqrtf(var + 1e-5f);
    #pragma unroll
    for (int j = 0; j < 4; ++j) {
        const int d = tid + j * 256;
        out[(size_t)b * DDIM + d] = (y[j] - mu) * rstd * gamma[d] + beta[d];
    }
}

__global__ void zero_kernel(float* __restrict__ p, int n) {
    const int i = blockIdx.x * blockDim.x + threadIdx.x;
    if (i < n) p[i] = 0.0f;
}

__global__ void loss_kernel(const float* __restrict__ lpart, float* __restrict__ out) {
    if (threadIdx.x == 0 && blockIdx.x == 0) {
        float aug = 0.f;
        for (int e = 0; e < NEXP; ++e)
            aug += (lpart[e] / (float)BTOK) * (lpart[NEXP + e] / (float)BTOK);
        aug /= (float)NEXP;
        const float s = lpart[2 * NEXP] / (float)BTOK;
        out[(size_t)BTOK * DDIM]     = aug;
        out[(size_t)BTOK * DDIM + 1] = (1.f - s) * (1.f - s);
    }
}

// ---------------------------------------------------------------------------
extern "C" void kernel_launch(void* const* d_in, const int* in_sizes, int n_in,
                              void* d_out, int out_size, void* d_ws, size_t ws_size,
                              hipStream_t stream)
{
    (void)in_sizes; (void)n_in; (void)out_size; (void)ws_size;
    const float* x      = (const float*)d_in[0];
    const float* logits = (const float*)d_in[1];
    const int*   masks  = (const int*)d_in[2];
    const float* ew_in  = (const float*)d_in[3];
    const float* ew_out = (const float*)d_in[4];
    const float* eb_out = (const float*)d_in[5];
    const float* gw_in  = (const float*)d_in[6];
    const float* gw_out = (const float*)d_in[7];
    const float* gb_out = (const float*)d_in[8];
    const float* ln_g   = (const float*)d_in[9];
    const float* ln_b   = (const float*)d_in[10];
    float* out = (float*)d_out;

    char* ws = (char*)d_ws;
    float*  accbuf = (float*)ws;                               // B*D f32 (8 MB)
    float*  gates  = (float*)(ws + (size_t)9 * 1024 * 1024);   // B*E f32
    float*  lpart  = gates + (size_t)BTOK * NEXP;              // 33 f32 (padded)
    bf16_t* H      = (bf16_t*)(ws + (size_t)16 * 1024 * 1024); // 18*B*F bf16 (288 MB)

    const int zn = BTOK * DDIM;
    zero_kernel<<<(zn + 255) / 256, 256, 0, stream>>>(accbuf, zn);
    zero_kernel<<<1, 64, 0, stream>>>(lpart, 64);
    gate_kernel<<<BTOK / 256, 256, 0, stream>>>(logits, masks, gates, lpart);
    fc1_gelu_kernel<<<dim3(FDIM / BN, BTOK / BM, NALL), 256, 0, stream>>>(x, ew_in, gw_in, H);
    fc2_combine_kernel<<<dim3(DDIM / BN, BTOK / BM, NALL), 256, 0, stream>>>(
        H, ew_out, eb_out, gw_out, gb_out, gates, accbuf);
    ln_kernel<<<BTOK, 256, 0, stream>>>(accbuf, x, ln_g, ln_b, out);
    loss_kernel<<<1, 32, 0, stream>>>(lpart, out);
}
